// Attention_14577119002882
// MI455X (gfx1250) — matrix operand order repaired and checked by reading, not compiled, not run on hardware
//
#include <hip/hip_runtime.h>

typedef __attribute__((ext_vector_type(16))) __bf16 v16bf;
typedef __attribute__((ext_vector_type(8)))  __bf16 v8bf;
typedef __attribute__((ext_vector_type(4)))  __bf16 v4bf;
typedef __attribute__((ext_vector_type(8)))  float  v8f;
typedef __attribute__((ext_vector_type(4)))  float  v4f;

#define EMB   1024
#define NH    16
#define HS    64
#define SEQ   2048
#define BATCH 4
#define N3    3072          // 3*EMB
#define MTOT  8192          // BATCH*SEQ

static __device__ __forceinline__ v8f wmma_bf16(v16bf a, v16bf b, v8f c) {
  return __builtin_amdgcn_wmma_f32_16x16x32_bf16(false, a, false, b, (short)0, c,
                                                 false, false);
}

static __device__ __forceinline__ void wait_asynccnt0() {
  asm volatile("s_wait_asynccnt 0" ::: "memory");
}

// one b128 async copy per lane: global -> LDS, tracked on ASYNCcnt
static __device__ __forceinline__ void async_b128_to_lds(const void* gaddr,
                                                         void* lds_generic) {
  unsigned lds_off = (unsigned)(uintptr_t)lds_generic;   // low 32 bits = LDS offset
  asm volatile("global_load_async_to_lds_b128 %0, %1, off"
               :
               : "v"(lds_off), "v"(gaddr)
               : "memory");
}

// ---------------------------------------------------------------- convert x
__global__ void __launch_bounds__(256) cvt_x_kernel(const float* __restrict__ x,
                                                    __bf16* __restrict__ xb, int n8) {
  int i = blockIdx.x * blockDim.x + threadIdx.x;
  if (i >= n8) return;
  const float4* p = (const float4*)x + (size_t)i * 2;
  float4 a = p[0], b = p[1];
  v8bf o;
  o[0] = (__bf16)a.x; o[1] = (__bf16)a.y; o[2] = (__bf16)a.z; o[3] = (__bf16)a.w;
  o[4] = (__bf16)b.x; o[5] = (__bf16)b.y; o[6] = (__bf16)b.z; o[7] = (__bf16)b.w;
  ((v8bf*)xb)[i] = o;
}

// ------------------------------------------- convert + transpose W -> [N3][EMB]
__global__ void __launch_bounds__(256) cvt_w_kernel(const float* __restrict__ W,
                                                    __bf16* __restrict__ Wt) {
  int i  = blockIdx.x * blockDim.x + threadIdx.x;   // one per 4 outputs
  int o4 = i * 4;
  int n  = o4 >> 10;          // / EMB
  int k0 = o4 & (EMB - 1);
  v4bf o;
#pragma unroll
  for (int j = 0; j < 4; ++j) o[j] = (__bf16)W[(size_t)(k0 + j) * N3 + n];
  *(v4bf*)(Wt + (size_t)n * EMB + k0) = o;
}

// ------------------------------------------------------------- QKV projection
// block tile 256M x 64N, wave tile 32M x 64N. Shared B tile (64 cols x 32 k)
// staged in LDS via async b128 copies, double buffered; k-loop fully unrolled
// so double-buffer rotation is pure register renaming (no movs, no hazard NOPs).
__global__ void __launch_bounds__(256) qkv_gemm_kernel(
    const __bf16* __restrict__ xb, const __bf16* __restrict__ Wt,
    const float* __restrict__ bias,
    __bf16* __restrict__ Qb, __bf16* __restrict__ Kb, __bf16* __restrict__ Vt) {
  __shared__ __bf16 Bt[2][64][32];     // 2 x 4KB

  int tid  = threadIdx.x;
  int wave = tid >> 5, lane = tid & 31;
  int lm = lane & 15, lk = lane >> 4;
  int mbase = blockIdx.x * 256 + wave * 32;
  int nbase = blockIdx.y * 64;

  // cooperative async copy of one 64x32 B tile: 256 threads x 16B
  int cc = tid >> 2;          // column 0..63
  int cp = tid & 3;           // 16B chunk 0..3
  const __bf16* bsrc_row = Wt + (size_t)(nbase + cc) * EMB + cp * 8;

  const __bf16* arow0 = xb + (size_t)(mbase + lm) * EMB;
  const __bf16* arow1 = arow0 + (size_t)16 * EMB;

  v8f vzero = {0.f, 0.f, 0.f, 0.f, 0.f, 0.f, 0.f, 0.f};
  v8f acc[2][4];
#pragma unroll
  for (int mt = 0; mt < 2; ++mt)
#pragma unroll
    for (int nt = 0; nt < 4; ++nt) acc[mt][nt] = vzero;

  // prologue: stage k-step 0 into buffer 0
  async_b128_to_lds(bsrc_row, &Bt[0][cc][cp * 8]);
  wait_asynccnt0();
  __syncthreads();

  // prologue: A fragments for k-step 0
  v16bf a_cur[2];
  {
    v8bf r0 = *(const v8bf*)(arow0 + lk * 8);
    v8bf r1 = *(const v8bf*)(arow0 + 16 + lk * 8);
    v8bf r2 = *(const v8bf*)(arow1 + lk * 8);
    v8bf r3 = *(const v8bf*)(arow1 + 16 + lk * 8);
#pragma unroll
    for (int i = 0; i < 8; ++i) {
      a_cur[0][i] = r0[i]; a_cur[0][8 + i] = r1[i];
      a_cur[1][i] = r2[i]; a_cur[1][8 + i] = r3[i];
    }
  }

#pragma unroll
  for (int ks = 0; ks < EMB / 32; ++ks) {            // 32 steps, fully unrolled
    const int kk = ks * 32;
    const int cur = ks & 1;
    const bool has_next = (ks + 1) < (EMB / 32);

    // issue async copy of next B tile into the other buffer (overlaps compute)
    if (has_next)
      async_b128_to_lds(bsrc_row + kk + 32, &Bt[cur ^ 1][cc][cp * 8]);

    // issue next A-fragment loads (overlap compute; renamed regs, no rotation)
    v16bf a_nxt[2];
    if (has_next) {
      v8bf r0 = *(const v8bf*)(arow0 + kk + 32 + lk * 8);
      v8bf r1 = *(const v8bf*)(arow0 + kk + 48 + lk * 8);
      v8bf r2 = *(const v8bf*)(arow1 + kk + 32 + lk * 8);
      v8bf r3 = *(const v8bf*)(arow1 + kk + 48 + lk * 8);
#pragma unroll
      for (int i = 0; i < 8; ++i) {
        a_nxt[0][i] = r0[i]; a_nxt[0][8 + i] = r1[i];
        a_nxt[1][i] = r2[i]; a_nxt[1][8 + i] = r3[i];
      }
    }

    // pull all 4 B fragments from LDS, then issue 8 back-to-back WMMAs
    v16bf bfr[4];
#pragma unroll
    for (int nt = 0; nt < 4; ++nt)
    {
      v8bf b0 = *(const v8bf*)&Bt[cur][nt * 16 + lm][lk * 8];
      v8bf b1 = *(const v8bf*)&Bt[cur][nt * 16 + lm][16 + lk * 8];
#pragma unroll
      for (int i = 0; i < 8; ++i) { bfr[nt][i] = b0[i]; bfr[nt][8 + i] = b1[i]; }
    }

#pragma unroll
    for (int mt = 0; mt < 2; ++mt)
#pragma unroll
      for (int nt = 0; nt < 4; ++nt)
        acc[mt][nt] = wmma_bf16(a_cur[mt], bfr[nt], acc[mt][nt]);

    if (has_next) { a_cur[0] = a_nxt[0]; a_cur[1] = a_nxt[1]; }

    wait_asynccnt0();        // own slice of next tile landed
    __syncthreads();         // everyone's slice landed; cur reads all consumed
  }

  // epilogue: bias + scatter to Q/K (bh,t,d) and Vt (bh,d,t); streaming stores
#pragma unroll
  for (int nt = 0; nt < 4; ++nt) {
    int ncol  = nbase + nt * 16 + lm;
    float bv  = bias[ncol];
    int h     = ncol / 192;
    int rem   = ncol - h * 192;
    int which = rem >> 6;       // 0=q 1=k 2=v
    int d     = rem & 63;
#pragma unroll
    for (int mt = 0; mt < 2; ++mt) {
#pragma unroll
      for (int r = 0; r < 8; ++r) {
        int m = mbase + mt * 16 + r + lk * 8;
        int b = m >> 11;           // / SEQ
        int t = m & (SEQ - 1);
        __bf16 v = (__bf16)(acc[mt][nt][r] + bv);
        size_t bh = (size_t)(b * NH + h);
        if (which == 0)
          __builtin_nontemporal_store(v, &Qb[(bh * SEQ + t) * HS + d]);
        else if (which == 1)
          __builtin_nontemporal_store(v, &Kb[(bh * SEQ + t) * HS + d]);
        else
          __builtin_nontemporal_store(v, &Vt[(bh * HS + d) * SEQ + t]);
      }
    }
  }
}

// -------------------------------------------------------------- attention
// one block per (b,h, 16-query-row tile); scores tile 16 x 2048 in LDS (fp32)
#define SMEM_FLOATS (16 * SEQ + 256 + 2 * 16 * 64)
#define SMEM_BYTES  (SMEM_FLOATS * 4)

__global__ void __launch_bounds__(256) attn_kernel(
    const __bf16* __restrict__ Qb, const __bf16* __restrict__ Kb,
    const __bf16* __restrict__ Vt, float* __restrict__ out) {
  extern __shared__ float smem[];
  float* S   = smem;               // [16][SEQ]
  float* red = smem + 16 * SEQ;    // [256]
  float* O2  = red + 256;          // [2][16][64]

  int tid  = threadIdx.x;
  int wave = tid >> 5, lane = tid & 31;
  int lm = lane & 15, lk = lane >> 4;
  int bh = blockIdx.x >> 7;        // 0..63
  int qt = blockIdx.x & 127;       // query tile

  const __bf16* qbase = Qb + ((size_t)bh * SEQ + qt * 16) * HS;
  const __bf16* kbase = Kb + (size_t)bh * SEQ * HS;
  const __bf16* vbase = Vt + (size_t)bh * HS * SEQ;

  v8f vzero = {0.f, 0.f, 0.f, 0.f, 0.f, 0.f, 0.f, 0.f};

  // --- Q fragments (2 K-chunks of 32 over hs=64), reused across all key tiles
  const __bf16* qrow = qbase + (size_t)lm * HS;
  v16bf qa[2];
#pragma unroll
  for (int c = 0; c < 2; ++c) {
    v8bf r0 = *(const v8bf*)(qrow + c * 32 + lk * 16);
    v8bf r1 = *(const v8bf*)(qrow + c * 32 + lk * 16 + 8);
#pragma unroll
    for (int i = 0; i < 8; ++i) { qa[c][i] = r0[i]; qa[c][8 + i] = r1[i]; }
  }

  // --- Phase 1: S = (Q @ K^T) * 1/sqrt(hs); 16 tiles per wave, unrolled so
  //     the K-fragment double buffer is register renaming
  {
    const __bf16* krow0 = kbase + (size_t)(wave * 16 + lm) * HS;
    v16bf kc0 = *(const v16bf*)(krow0 + lk * 16);
    v16bf kc1 = *(const v16bf*)(krow0 + 32 + lk * 16);
#pragma unroll
    for (int it = 0; it < 16; ++it) {
      int nt = wave + it * 8;
      v16bf kn0 = kc0, kn1 = kc1;
      if (it + 1 < 16) {
        const __bf16* krn = kbase + (size_t)((nt + 8) * 16 + lm) * HS;
        kn0 = *(const v16bf*)(krn + lk * 16);
        kn1 = *(const v16bf*)(krn + 32 + lk * 16);
      }
      v8f acc = vzero;
      acc = wmma_bf16(qa[0], kc0, acc);
      acc = wmma_bf16(qa[1], kc1, acc);
#pragma unroll
      for (int r = 0; r < 8; ++r)
        S[(size_t)(r + lk * 8) * SEQ + nt * 16 + lm] = acc[r] * 0.125f;
      kc0 = kn0; kc1 = kn1;
    }
  }
  __syncthreads();

  // --- Phase 2: softmax, float4-vectorized (16 threads per row, 32 vec4 each)
  {
    int row = tid >> 4, j0 = tid & 15;
    float* srow = S + (size_t)row * SEQ;
    float mx = -3.402823466e38f;
#pragma unroll 4
    for (int i = 0; i < SEQ / 64; ++i) {
      v4f v = *(const v4f*)(srow + j0 * 4 + i * 64);
      mx = fmaxf(mx, fmaxf(fmaxf(v[0], v[1]), fmaxf(v[2], v[3])));
    }
    red[tid] = mx;
    __syncthreads();
#pragma unroll
    for (int k = 0; k < 16; ++k) mx = fmaxf(mx, red[row * 16 + k]);
    __syncthreads();
    float sum = 0.f;
#pragma unroll 4
    for (int i = 0; i < SEQ / 64; ++i) {
      float* p = srow + j0 * 4 + i * 64;
      v4f v = *(const v4f*)p;
      v4f e;
      e[0] = __expf(v[0] - mx); e[1] = __expf(v[1] - mx);
      e[2] = __expf(v[2] - mx); e[3] = __expf(v[3] - mx);
      *(v4f*)p = e;
      sum += e[0] + e[1] + e[2] + e[3];
    }
    red[tid] = sum;
    __syncthreads();
    float tot = 0.f;
#pragma unroll
    for (int k = 0; k < 16; ++k) tot += red[row * 16 + k];
    float inv = 1.f / tot;
#pragma unroll 4
    for (int i = 0; i < SEQ / 64; ++i) {
      float* p = srow + j0 * 4 + i * 64;
      v4f v = *(const v4f*)p;
      v[0] *= inv; v[1] *= inv; v[2] *= inv; v[3] *= inv;
      *(v4f*)p = v;
    }
  }
  __syncthreads();

  // --- Phase 3: out_tile = P(16xSEQ) @ V(SEQ x 64); split-K over wave pairs
  {
    int half = wave >> 2;            // K half: 0 -> [0,1024), 1 -> [1024,2048)
    int nt   = wave & 3;             // 16-wide d tile
    int kb0  = half * 1024;
    const float*  prow = S + (size_t)lm * SEQ + kb0;
    const __bf16* vrow = vbase + (size_t)(nt * 16 + lm) * SEQ + kb0;

    v16bf vc = *(const v16bf*)(vrow + lk * 16);
    v8f acc = vzero;
#pragma unroll 4
    for (int it = 0; it < 32; ++it) {
      int kk = it * 32;
      v16bf vn = vc;
      if (it + 1 < 32)
        vn = *(const v16bf*)(vrow + kk + 32 + lk * 16);
      v4f p0 = *(const v4f*)(prow + kk + lk * 16);
      v4f p1 = *(const v4f*)(prow + kk + lk * 16 + 4);
      v4f p2 = *(const v4f*)(prow + kk + lk * 16 + 8);
      v4f p3 = *(const v4f*)(prow + kk + lk * 16 + 12);
      v16bf pa;
#pragma unroll
      for (int i = 0; i < 4; ++i) {
        pa[i]      = (__bf16)p0[i];
        pa[4 + i]  = (__bf16)p1[i];
        pa[8 + i]  = (__bf16)p2[i];
        pa[12 + i] = (__bf16)p3[i];
      }
      acc = wmma_bf16(pa, vc, acc);
      vc = vn;
    }
#pragma unroll
    for (int r = 0; r < 8; ++r)
      O2[half * 1024 + (r + lk * 8) * 64 + nt * 16 + lm] = acc[r];
  }
  __syncthreads();

  // --- deterministic partial reduction + streaming store [b, t, h*hs + d]
  {
    int b = bh >> 4, h = bh & 15;
    int idx = tid * 4;
    int m = idx >> 6, d0 = idx & 63;
    v4f o;
    o[0] = O2[idx + 0] + O2[1024 + idx + 0];
    o[1] = O2[idx + 1] + O2[1024 + idx + 1];
    o[2] = O2[idx + 2] + O2[1024 + idx + 2];
    o[3] = O2[idx + 3] + O2[1024 + idx + 3];
    size_t off = ((size_t)b * SEQ + qt * 16 + m) * EMB + h * HS + d0;
    __builtin_nontemporal_store(o, (v4f*)(out + off));
  }
}

// ------------------------------------------------------------------ launcher
extern "C" void kernel_launch(void* const* d_in, const int* in_sizes, int n_in,
                              void* d_out, int out_size, void* d_ws, size_t ws_size,
                              hipStream_t stream) {
  const float* x  = (const float*)d_in[0];
  const float* W  = (const float*)d_in[1];
  const float* bq = (const float*)d_in[2];
  float* out = (float*)d_out;

  char* ws = (char*)d_ws;
  __bf16* xb = (__bf16*)ws;  ws += (size_t)MTOT * EMB * 2;            // 16 MB
  __bf16* Wt = (__bf16*)ws;  ws += (size_t)N3 * EMB * 2;              //  6 MB
  __bf16* Qb = (__bf16*)ws;  ws += (size_t)BATCH * NH * SEQ * HS * 2; // 16 MB
  __bf16* Kb = (__bf16*)ws;  ws += (size_t)BATCH * NH * SEQ * HS * 2; // 16 MB
  __bf16* Vt = (__bf16*)ws;                                           // 16 MB

  (void)in_sizes; (void)n_in; (void)out_size; (void)ws_size;

  hipFuncSetAttribute(reinterpret_cast<const void*>(attn_kernel),
                      hipFuncAttributeMaxDynamicSharedMemorySize, SMEM_BYTES);

  int n8 = (MTOT * EMB) / 8;                        // 1048576
  cvt_x_kernel<<<n8 / 256, 256, 0, stream>>>(x, xb, n8);
  cvt_w_kernel<<<(N3 * EMB / 4) / 256, 256, 0, stream>>>(W, Wt);
  qkv_gemm_kernel<<<dim3(MTOT / 256, N3 / 64), 256, 0, stream>>>(xb, Wt, bq,
                                                                 Qb, Kb, Vt);
  attn_kernel<<<BATCH * NH * (SEQ / 16), 256, SMEM_BYTES, stream>>>(Qb, Kb, Vt,
                                                                    out);
}
